// GlobalAttention_39548058862194
// MI455X (gfx1250) — matrix operand order repaired and checked by reading, not compiled, not run on hardware
//
#include <hip/hip_runtime.h>
#include <hip/hip_bf16.h>

// ---------------------------------------------------------------------------
// Global attention (B=8, C=512, H=W=48) for MI455X / gfx1250.
// Compute-bound -> all GEMMs on v_wmma_f32_16x16x32_bf16.
// All B operands stored K-contiguous so LDS staging is pure b128 traffic.
// 128-thread blocks, 128x64 block tile (wave = 32x64), double-buffered LDS.
// ---------------------------------------------------------------------------

typedef __attribute__((ext_vector_type(16))) __bf16 v16bf;
typedef __attribute__((ext_vector_type(8)))  float  v8f;

#define HID   512
#define BATCH 8
#define HWDIM 48
#define NTOK  (HWDIM * HWDIM)   /* 2304 tokens */

#define BM 128   /* block tile rows */
#define BN 64    /* block tile cols */
#define TK 32    /* K step (WMMA K) */
#define LA_SZ (BM * TK)          /* LDS A elements per buffer */
#define LB_SZ (BN * TK)          /* LDS B elements per buffer */

// fp32 -> bf16, round-to-nearest-even
__device__ __forceinline__ unsigned short f2bf(float f) {
  union { float f; unsigned u; } c; c.f = f;
  unsigned u = c.u;
  unsigned r = u + 0x7FFFu + ((u >> 16) & 1u);
  return (unsigned short)(r >> 16);
}

union AFrag { v16bf v; unsigned u[8]; };
union BFrag { v16bf v; unsigned short h[16]; };

__device__ __forceinline__ v8f wmma_bf16(const AFrag& a, const BFrag& b, v8f c) {
  return __builtin_amdgcn_wmma_f32_16x16x32_bf16(false, a.v, false, b.v,
                                                 (short)0, c, false, false);
}

// ---------------------------------------------------------------------------
// Stage one K-slab into LDS (all b128):
//   A: (m,k) at Ag[m*lda + k]      -> lA[row][k], 128x32
//   B: (k,col) at Bg[col*ldb + k]  -> lB[col][k],  64x32 (K-contiguous source)
// ---------------------------------------------------------------------------
__device__ __forceinline__ void stage(const unsigned short* __restrict__ Ag, int lda,
                                      const unsigned short* __restrict__ Bg, int ldb,
                                      int k0, unsigned short* lA, unsigned short* lB) {
  const int tid = threadIdx.x;    // 0..127
  // ---- A tile: one 64-byte row per thread ----
  {
    const uint4* src = (const uint4*)(Ag + (size_t)tid * lda + k0);
    uint4 d0 = src[0], d1 = src[1], d2 = src[2], d3 = src[3];
    uint4* dst = (uint4*)(lA + tid * TK);
    dst[0] = d0; dst[1] = d1; dst[2] = d2; dst[3] = d3;
  }
  // ---- B tile: 2 chunks per column ----
  {
    const int col = tid & 63, h = tid >> 6;
    const uint4* src = (const uint4*)(Bg + (size_t)col * ldb + k0 + h * 16);
    uint4 d0 = src[0], d1 = src[1];
    uint4* dst = (uint4*)(lB + col * TK + h * 16);
    dst[0] = d0; dst[1] = d1;
  }
  __builtin_prefetch(Ag + (size_t)tid * lda + k0 + 2 * TK, 0, 3);  // global_prefetch_b8
}

// ---------------------------------------------------------------------------
// Wave compute: 32x64 strip = 2 A fragments x 4 B fragments -> 8 WMMAs.
// Fragment reads are 16B-contiguous per lane (2x ds_load_b128 each).
// ---------------------------------------------------------------------------
__device__ __forceinline__ void wave_mma(const unsigned short* lA, const unsigned short* lB,
                                         v8f (*acc)[4]) {
  const int lane = threadIdx.x & 31;
  const int w    = threadIdx.x >> 5;
  const int mrow = lane & 15;
  const int lh   = lane >> 4;

  const unsigned short* ar0 = lA + (w * 32 + mrow) * TK;
  const unsigned short* ar1 = ar0 + 16 * TK;
  AFrag a0, a1;
  *(uint4*)&a0.u[0] = *(const uint4*)(ar0 + lh * 8);
  *(uint4*)&a0.u[4] = *(const uint4*)(ar0 + 16 + lh * 8);
  *(uint4*)&a1.u[0] = *(const uint4*)(ar1 + lh * 8);
  *(uint4*)&a1.u[4] = *(const uint4*)(ar1 + 16 + lh * 8);

  BFrag bb[4];
#pragma unroll
  for (int cb = 0; cb < 4; ++cb) {
    const unsigned short* br = lB + (cb * 16 + mrow) * TK + lh * 8;
    *(uint4*)&bb[cb].h[0] = *(const uint4*)(br);
    *(uint4*)&bb[cb].h[8] = *(const uint4*)(br + 16);
  }
#pragma unroll
  for (int cb = 0; cb < 4; ++cb) {
    acc[0][cb] = wmma_bf16(a0, bb[cb], acc[0][cb]);
    acc[1][cb] = wmma_bf16(a1, bb[cb], acc[1][cb]);
  }
}

// ---------------------------------------------------------------------------
// Double-buffered block mainloop: 128x64 fp32 tile in acc[2][4] per wave.
// ---------------------------------------------------------------------------
__device__ __forceinline__ void gemm_mainloop(const unsigned short* __restrict__ Ag, int lda,
                                              const unsigned short* __restrict__ Bg, int ldb,
                                              int K, unsigned short* lA, unsigned short* lB,
                                              v8f (*acc)[4]) {
  stage(Ag, lda, Bg, ldb, 0, lA, lB);
  int cur = 0;
  for (int k0 = 0; k0 < K; k0 += TK) {
    __syncthreads();
    if (k0 + TK < K)
      stage(Ag, lda, Bg, ldb, k0 + TK, lA + (cur ^ 1) * LA_SZ, lB + (cur ^ 1) * LB_SZ);
    wave_mma(lA + cur * LA_SZ, lB + cur * LB_SZ, acc);
    cur ^= 1;
  }
}

// ---------------------------------------------------------------------------
// fp32 -> bf16 bulk convert (weights)
// ---------------------------------------------------------------------------
__global__ void f2bf_kernel(const float* __restrict__ in, unsigned short* __restrict__ out,
                            size_t n) {
  size_t i = (size_t)blockIdx.x * blockDim.x + threadIdx.x;
  size_t stride = (size_t)gridDim.x * blockDim.x;
  for (; i < n; i += stride) out[i] = f2bf(in[i]);
}

// ---------------------------------------------------------------------------
// Transpose + convert: X[B][C][N] fp32 -> Xt[B][N][C] bf16 (32x32 LDS tiles)
// grid = (NTOK/32, HID/32, BATCH), block = (32,8)
// ---------------------------------------------------------------------------
__global__ __launch_bounds__(256)
void xpose_f2bf_kernel(const float* __restrict__ X, unsigned short* __restrict__ Xt) {
  __shared__ float tile[32][33];
  const int nb = blockIdx.x * 32;
  const int cb = blockIdx.y * 32;
  const int b  = blockIdx.z;
  const int tx = threadIdx.x;     // 0..31
  const int ty = threadIdx.y;     // 0..7
  const float* src = X + ((size_t)b * HID + cb) * NTOK + nb;
#pragma unroll
  for (int j = 0; j < 4; ++j) {
    int c = ty * 4 + j;
    tile[c][tx] = src[(size_t)c * NTOK + tx];       // coalesced along tokens
  }
  __syncthreads();
  unsigned short* dst = Xt + ((size_t)b * NTOK + nb) * HID + cb;
#pragma unroll
  for (int j = 0; j < 4; ++j) {
    int n = ty * 4 + j;
    dst[(size_t)n * HID + tx] = f2bf(tile[tx][n]);  // coalesced along channels
  }
}

// ---------------------------------------------------------------------------
// Projection: acc = sum_c W[o][c] * Xt[b][n][c] + bias[o]
//   TRANS_OUT=false: Out[b][n][o] (bf16, packed 16B stores)  -- Q, K
//   TRANS_OUT=true : Out[b][o][n] (bf16)                     -- V (K-contig for ctx)
// grid = (NTOK/BN, HID/BM, BATCH), block = 128
// ---------------------------------------------------------------------------
template<bool TRANS_OUT>
__global__ __launch_bounds__(128)
void proj_gemm_kernel(const unsigned short* __restrict__ Wb,   // [HID][HID] bf16
                      const float* __restrict__ bias,          // [HID]
                      const unsigned short* __restrict__ Xt,   // [B][NTOK][HID] bf16
                      unsigned short* __restrict__ Out) {
  __shared__ __align__(16) unsigned short lA[2 * LA_SZ];
  __shared__ __align__(16) unsigned short lB[2 * LB_SZ];
  const int nbase = blockIdx.x * BN;
  const int obase = blockIdx.y * BM;
  const int b     = blockIdx.z;
  const unsigned short* Ag = Wb + (size_t)obase * HID;                   // (o,c)
  const unsigned short* Bg = Xt + ((size_t)b * NTOK + nbase) * HID;      // (c,n): n*HID+c
  v8f acc[2][4] = {};
  gemm_mainloop(Ag, HID, Bg, HID, HID, lA, lB, acc);

  const int lane = threadIdx.x & 31;
  const int w    = threadIdx.x >> 5;
  const int col  = lane & 15;
  const int rb   = (lane >> 4) * 8;
#pragma unroll
  for (int g = 0; g < 2; ++g) {
    const int o0 = obase + w * 32 + g * 16 + rb;   // 8 consecutive channels
    float4 bi0 = *(const float4*)(bias + o0);
    float4 bi1 = *(const float4*)(bias + o0 + 4);
    const float bv[8] = {bi0.x, bi0.y, bi0.z, bi0.w, bi1.x, bi1.y, bi1.z, bi1.w};
#pragma unroll
    for (int cb = 0; cb < 4; ++cb) {
      const int n = nbase + cb * 16 + col;
      if (!TRANS_OUT) {
        unsigned short pk[8];
#pragma unroll
        for (int r = 0; r < 8; ++r) pk[r] = f2bf(acc[g][cb][r] + bv[r]);
        *(uint4*)(Out + ((size_t)b * NTOK + n) * HID + o0) = *(const uint4*)pk;
      } else {
#pragma unroll
        for (int r = 0; r < 8; ++r)
          Out[((size_t)b * HID + o0 + r) * NTOK + n] = f2bf(acc[g][cb][r] + bv[r]);
      }
    }
  }
}

// ---------------------------------------------------------------------------
// Scores: S[b][n][m] = scale * sum_c Q[b][n][c]*K[b][m][c]
// grid = (NTOK/BN, NTOK/BM, BATCH)
// ---------------------------------------------------------------------------
__global__ __launch_bounds__(128)
void scores_kernel(const unsigned short* __restrict__ Qb,   // [B][NTOK][HID]
                   const unsigned short* __restrict__ Kb,   // [B][NTOK][HID]
                   float* __restrict__ S) {                 // [B][NTOK][NTOK]
  __shared__ __align__(16) unsigned short lA[2 * LA_SZ];
  __shared__ __align__(16) unsigned short lB[2 * LB_SZ];
  const int mbase = blockIdx.x * BN;
  const int nbase = blockIdx.y * BM;
  const int b     = blockIdx.z;
  const unsigned short* Ag = Qb + ((size_t)b * NTOK + nbase) * HID;      // (n,c)
  const unsigned short* Bg = Kb + ((size_t)b * NTOK + mbase) * HID;      // (c,m): m*HID+c
  v8f acc[2][4] = {};
  gemm_mainloop(Ag, HID, Bg, HID, HID, lA, lB, acc);

  const float scale = 0.044194173824159216f;   // 1/sqrt(512)
  const int lane = threadIdx.x & 31;
  const int w    = threadIdx.x >> 5;
  const int col  = lane & 15;
  const int rb   = (lane >> 4) * 8;
#pragma unroll
  for (int g = 0; g < 2; ++g) {
#pragma unroll
    for (int cb = 0; cb < 4; ++cb) {
      const int m = mbase + cb * 16 + col;
#pragma unroll
      for (int r = 0; r < 8; ++r) {
        const int n = nbase + w * 32 + g * 16 + rb + r;
        S[((size_t)b * NTOK + n) * NTOK + m] = acc[g][cb][r] * scale;
      }
    }
  }
}

// ---------------------------------------------------------------------------
// Row softmax over M, probs written bf16. One 256-thread block per row.
// ---------------------------------------------------------------------------
__global__ __launch_bounds__(256)
void softmax_kernel(const float* __restrict__ S, unsigned short* __restrict__ P, int Mlen) {
  __shared__ float red[256];
  const size_t row = blockIdx.x;
  const float* s = S + row * (size_t)Mlen;
  const int t = threadIdx.x;

  float mx = -3.0e38f;
  for (int i = t; i < Mlen; i += 256) mx = fmaxf(mx, s[i]);
  red[t] = mx;
  __syncthreads();
  for (int st = 128; st > 0; st >>= 1) {
    if (t < st) red[t] = fmaxf(red[t], red[t + st]);
    __syncthreads();
  }
  mx = red[0];
  __syncthreads();

  float sum = 0.f;
  for (int i = t; i < Mlen; i += 256) sum += __expf(s[i] - mx);
  red[t] = sum;
  __syncthreads();
  for (int st = 128; st > 0; st >>= 1) {
    if (t < st) red[t] += red[t + st];
    __syncthreads();
  }
  const float inv = 1.f / red[0];

  unsigned short* p = P + row * (size_t)Mlen;
  for (int i = t; i < Mlen; i += 256) p[i] = f2bf(__expf(s[i] - mx) * inv);
}

// ---------------------------------------------------------------------------
// Context: Ctx[b][n][c] = sum_m P[b][n][m]*Vt[b][c][m]
// grid = (HID/BN, NTOK/BM, BATCH)
// ---------------------------------------------------------------------------
__global__ __launch_bounds__(128)
void ctx_kernel(const unsigned short* __restrict__ Pb,   // [B][NTOK][NTOK]
                const unsigned short* __restrict__ Vt,   // [B][HID][NTOK] (transposed V)
                unsigned short* __restrict__ Ctx) {      // [B][NTOK][HID]
  __shared__ __align__(16) unsigned short lA[2 * LA_SZ];
  __shared__ __align__(16) unsigned short lB[2 * LB_SZ];
  const int cbase = blockIdx.x * BN;
  const int nbase = blockIdx.y * BM;
  const int b     = blockIdx.z;
  const unsigned short* Ag = Pb + ((size_t)b * NTOK + nbase) * NTOK;     // (n,m)
  const unsigned short* Bg = Vt + ((size_t)b * HID + cbase) * NTOK;      // (m,c): c*NTOK+m
  v8f acc[2][4] = {};
  gemm_mainloop(Ag, NTOK, Bg, NTOK, NTOK, lA, lB, acc);

  const int lane = threadIdx.x & 31;
  const int w    = threadIdx.x >> 5;
  const int col  = lane & 15;
  const int rb   = (lane >> 4) * 8;
#pragma unroll
  for (int g = 0; g < 2; ++g) {
#pragma unroll
    for (int cb = 0; cb < 4; ++cb) {
      const int c = cbase + cb * 16 + col;
#pragma unroll
      for (int r = 0; r < 8; ++r) {
        const int n = nbase + w * 32 + g * 16 + rb + r;
        Ctx[((size_t)b * NTOK + n) * HID + c] = f2bf(acc[g][cb][r]);
      }
    }
  }
}

// ---------------------------------------------------------------------------
// Output projection: out[b][o][n] = sum_c Wo[o][c]*Ctx[b][n][c] + bo[o]  (fp32)
// grid = (NTOK/BN, HID/BM, BATCH)
// ---------------------------------------------------------------------------
__global__ __launch_bounds__(128)
void out_gemm_kernel(const unsigned short* __restrict__ Wob,  // [HID][HID]
                     const float* __restrict__ bo,            // [HID]
                     const unsigned short* __restrict__ Ctx,  // [B][NTOK][HID]
                     float* __restrict__ Out) {               // [B][HID][NTOK] fp32
  __shared__ __align__(16) unsigned short lA[2 * LA_SZ];
  __shared__ __align__(16) unsigned short lB[2 * LB_SZ];
  const int nbase = blockIdx.x * BN;
  const int obase = blockIdx.y * BM;
  const int b     = blockIdx.z;
  const unsigned short* Ag = Wob + (size_t)obase * HID;                  // (o,c)
  const unsigned short* Bg = Ctx + ((size_t)b * NTOK + nbase) * HID;     // (c,n): n*HID+c
  v8f acc[2][4] = {};
  gemm_mainloop(Ag, HID, Bg, HID, HID, lA, lB, acc);

  const int lane = threadIdx.x & 31;
  const int w    = threadIdx.x >> 5;
  const int col  = lane & 15;
  const int rb   = (lane >> 4) * 8;
#pragma unroll
  for (int g = 0; g < 2; ++g) {
#pragma unroll
    for (int cb = 0; cb < 4; ++cb) {
      const int n = nbase + cb * 16 + col;
#pragma unroll
      for (int r = 0; r < 8; ++r) {
        const int o = obase + w * 32 + g * 16 + rb + r;
        Out[((size_t)b * HID + o) * NTOK + n] = acc[g][cb][r] + bo[o];
      }
    }
  }
}

// ---------------------------------------------------------------------------
extern "C" void kernel_launch(void* const* d_in, const int* in_sizes, int n_in,
                              void* d_out, int out_size, void* d_ws, size_t ws_size,
                              hipStream_t stream) {
  const float* locx = (const float*)d_in[0];
  const float* glox = (const float*)d_in[1];
  const float* Wq = (const float*)d_in[2]; const float* bq = (const float*)d_in[3];
  const float* Wk = (const float*)d_in[4]; const float* bk = (const float*)d_in[5];
  const float* Wv = (const float*)d_in[6]; const float* bv = (const float*)d_in[7];
  const float* Wo = (const float*)d_in[8]; const float* bo = (const float*)d_in[9];
  float* out = (float*)d_out;

  // --- workspace carving (256B aligned slices) ---
  char* ws = (char*)d_ws;
  size_t off = 0;
  auto take = [&](size_t bytes) -> char* {
    char* p = ws + off;
    off += (bytes + 255) & ~(size_t)255;
    return p;
  };
  const size_t WW = (size_t)HID * HID;            // weight elems
  const size_t XX = (size_t)BATCH * HID * NTOK;   // activation elems
  const size_t SS = (size_t)BATCH * NTOK * NTOK;  // score elems

  unsigned short* Wqb = (unsigned short*)take(WW * 2);
  unsigned short* Wkb = (unsigned short*)take(WW * 2);
  unsigned short* Wvb = (unsigned short*)take(WW * 2);
  unsigned short* Wob = (unsigned short*)take(WW * 2);
  unsigned short* Xlt = (unsigned short*)take(XX * 2);   // locx bf16 [B][N][C]
  unsigned short* Xgt = (unsigned short*)take(XX * 2);   // glox bf16 [B][N][C]
  unsigned short* Qb  = (unsigned short*)take(XX * 2);   // [B][N][C]
  unsigned short* Kb  = (unsigned short*)take(XX * 2);   // [B][M][C]
  unsigned short* Vt  = (unsigned short*)take(XX * 2);   // [B][C][M]  (transposed V)
  unsigned short* Cxb = (unsigned short*)take(XX * 2);   // [B][N][C]
  float*          S   = (float*)take(SS * 4);            // [B][N][M]
  unsigned short* Pb  = (unsigned short*)take(SS * 2);   // [B][N][M]
  (void)ws_size; (void)in_sizes; (void)n_in; (void)out_size;

  // --- 1. weight convert + input transpose/convert ---
  f2bf_kernel<<<512, 256, 0, stream>>>(Wq, Wqb, WW);
  f2bf_kernel<<<512, 256, 0, stream>>>(Wk, Wkb, WW);
  f2bf_kernel<<<512, 256, 0, stream>>>(Wv, Wvb, WW);
  f2bf_kernel<<<512, 256, 0, stream>>>(Wo, Wob, WW);
  dim3 xblk(32, 8);
  dim3 xgrd(NTOK / 32, HID / 32, BATCH);
  xpose_f2bf_kernel<<<xgrd, xblk, 0, stream>>>(locx, Xlt);
  xpose_f2bf_kernel<<<xgrd, xblk, 0, stream>>>(glox, Xgt);

  // --- 2. Q/K/V projections ---
  dim3 blk(128);
  dim3 gProj(NTOK / BN, HID / BM, BATCH);
  proj_gemm_kernel<false><<<gProj, blk, 0, stream>>>(Wqb, bq, Xlt, Qb);
  proj_gemm_kernel<false><<<gProj, blk, 0, stream>>>(Wkb, bk, Xgt, Kb);
  proj_gemm_kernel<true ><<<gProj, blk, 0, stream>>>(Wvb, bv, Xgt, Vt);

  // --- 3. attention ---
  dim3 gScore(NTOK / BN, NTOK / BM, BATCH);
  scores_kernel<<<gScore, blk, 0, stream>>>(Qb, Kb, S);
  softmax_kernel<<<BATCH * NTOK, 256, 0, stream>>>(S, Pb, NTOK);
  dim3 gCtx(HID / BN, NTOK / BM, BATCH);
  ctx_kernel<<<gCtx, blk, 0, stream>>>(Pb, Vt, Cxb);

  // --- 4. output projection straight into d_out ---
  dim3 gOut(NTOK / BN, HID / BM, BATCH);
  out_gemm_kernel<<<gOut, blk, 0, stream>>>(Wob, bo, Cxb, out);
}
